// SDformerVQVAE_1047972020233
// MI455X (gfx1250) — matrix operand order repaired and checked by reading, not compiled, not run on hardware
//
#include <hip/hip_runtime.h>
#include <hip/hip_bf16.h>

typedef __attribute__((ext_vector_type(16))) __bf16       v16bf;
typedef __attribute__((ext_vector_type(8)))  float        v8f;
typedef __attribute__((ext_vector_type(4)))  unsigned int u32x4;
typedef __attribute__((ext_vector_type(8)))  unsigned int u32x8;

// ---------------------------------------------------------------------------
// Implicit-GEMM 1D conv, (B,T,C) bf16 activations, fp32 WMMA accumulate.
// One wave32 computes a 16(M=positions) x 16*NT(N=out channels) tile.
// A frag (16-bit A 16x32): lane -> row M=lane&15; K groups
//   [kk0+16*half, +8) -> vec elems 0..7,  [kk0+16*half+8, +8) -> elems 8..15.
// B frag (16-bit B 32x16): lane -> col N=lane&15; K = [kk0+16*half, +16).
// C/D: vgpr v, lane L -> (M = v + 8*(L>=16), N = L&15).
// Fast path (Cin >= 32): K loop = taps(kw) x channel chunks; per-tap tin,
// validity and base pointer hoisted -> inner loop is pure const-offset b128
// loads + cndmask + 2 WMMAs.  Generic path (Cin==16) uses K padded to 32.
// ---------------------------------------------------------------------------
template <int NT>
__global__ __launch_bounds__(256) void conv1d_btc_wmma(
    const __bf16* __restrict__ in,     // (B, TinStore, Cin)
    const __bf16* __restrict__ wt,     // (Cout, KtotPad) row-major, kk = kw*Cin+ci
    const float*  __restrict__ bias,   // (Cout)
    const __bf16* __restrict__ resid,  // (B, Tout, Cout) or null
    __bf16* __restrict__ outb,         // mode 0: (B, Tout, Cout) bf16
    float*  __restrict__ outf,         // mode 1: (B, Cout, Tout) fp32 (NCH)
    int TinStore, int TinEff, int Tout,
    int cinLog2, int Cout, int Kw, int KtotPad, int stride, int pad, int dil,
    int halveIn, int reluFlag, int mode, int Ntiles, int totalWaves)
{
  int wave = (int)((blockIdx.x * blockDim.x + threadIdx.x) >> 5);
  if (wave >= totalWaves) return;            // wave-uniform: EXEC stays all-1s
  int lane  = threadIdx.x & 31;
  int half  = lane >> 4;
  int l16   = lane & 15;
  int tileN = wave % Ntiles;
  int tileM = wave / Ntiles;
  int Cin   = 1 << cinLog2;
  int cmask = Cin - 1;
  int n0    = tileN * (16 * NT) + l16;
  int mA    = tileM * 16 + l16;
  int tb    = mA / Tout;
  int tt    = mA - tb * Tout;
  const __bf16* inB = in + (size_t)tb * TinStore * Cin;
  const u32x4* wrow0 = (const u32x4*)(wt + (size_t)n0 * KtotPad);
  const u32x4* wrow1 = (const u32x4*)(wt + (size_t)(n0 + 16) * KtotPad);
  int tbase = tt * stride - pad;
  int maxTs = TinStore - 1;
  const u32x4 zq = {0u, 0u, 0u, 0u};

  v8f acc[NT];
#pragma unroll
  for (int u = 0; u < NT; ++u) acc[u] = (v8f){0.f,0.f,0.f,0.f,0.f,0.f,0.f,0.f};

  if (cinLog2 >= 5) {
    // ---------------- fast path: per-tap hoisted addressing ----------------
    int qb = (half << 1);                       // u32x4 index into weight rows
    for (int kw = 0; kw < Kw; ++kw) {
      int tin = tbase + kw * dil;
      bool ok = (tin >= 0) & (tin < TinEff);
      int ts  = halveIn ? (tin >> 1) : tin;
      ts = ts < 0 ? 0 : (ts > maxTs ? maxTs : ts);
      const u32x4* abase =
          (const u32x4*)(inB + (size_t)ts * Cin + (half << 4));
#pragma unroll 2
      for (int ci0 = 0; ci0 < Cin; ci0 += 32, qb += 4) {
        int qa = ci0 >> 3;
        u32x4 g0 = abase[qa];
        u32x4 g1 = abase[qa + 1];
        g0 = ok ? g0 : zq;
        g1 = ok ? g1 : zq;
        u32x8 au = __builtin_shufflevector(g0, g1, 0, 1, 2, 3, 4, 5, 6, 7);
        v16bf av = __builtin_bit_cast(v16bf, au);
        {
          u32x4 h0 = wrow0[qb], h1 = wrow0[qb + 1];
          u32x8 bu = __builtin_shufflevector(h0, h1, 0, 1, 2, 3, 4, 5, 6, 7);
          v16bf bv = __builtin_bit_cast(v16bf, bu);
          acc[0] = __builtin_amdgcn_wmma_f32_16x16x32_bf16(
              false, av, false, bv, (short)0, acc[0], false, false);
        }
        if constexpr (NT == 2) {
          u32x4 h0 = wrow1[qb], h1 = wrow1[qb + 1];
          u32x8 bu = __builtin_shufflevector(h0, h1, 0, 1, 2, 3, 4, 5, 6, 7);
          v16bf bv = __builtin_bit_cast(v16bf, bu);
          acc[1] = __builtin_amdgcn_wmma_f32_16x16x32_bf16(
              false, av, false, bv, (short)0, acc[1], false, false);
        }
      }
    }
  } else {
    // ---------------- generic path (Cin == 16, K padded) ----------------
    auto loadA = [&](int s) -> u32x4 {
      int kw  = s >> cinLog2;
      int ci  = s & cmask;
      int tin = tbase + kw * dil;
      bool ok = (tin >= 0) & (tin < TinEff);
      int ts  = halveIn ? (tin >> 1) : tin;
      ts = ts < 0 ? 0 : (ts > maxTs ? maxTs : ts);
      u32x4 val = *(const u32x4*)(inB + (size_t)ts * Cin + ci);
      return ok ? val : zq;
    };
    for (int kk0 = 0; kk0 < KtotPad; kk0 += 32) {
      int s0 = kk0 + (half << 4);
      u32x4 g0 = loadA(s0);
      u32x4 g1 = loadA(s0 + 8);
      u32x8 au = __builtin_shufflevector(g0, g1, 0, 1, 2, 3, 4, 5, 6, 7);
      v16bf av = __builtin_bit_cast(v16bf, au);
      int q = (kk0 + (half << 4)) >> 3;
      {
        u32x4 h0 = wrow0[q], h1 = wrow0[q + 1];
        u32x8 bu = __builtin_shufflevector(h0, h1, 0, 1, 2, 3, 4, 5, 6, 7);
        v16bf bv = __builtin_bit_cast(v16bf, bu);
        acc[0] = __builtin_amdgcn_wmma_f32_16x16x32_bf16(
            false, av, false, bv, (short)0, acc[0], false, false);
      }
      if constexpr (NT == 2) {
        u32x4 h0 = wrow1[q], h1 = wrow1[q + 1];
        u32x8 bu = __builtin_shufflevector(h0, h1, 0, 1, 2, 3, 4, 5, 6, 7);
        v16bf bv = __builtin_bit_cast(v16bf, bu);
        acc[1] = __builtin_amdgcn_wmma_f32_16x16x32_bf16(
            false, av, false, bv, (short)0, acc[1], false, false);
      }
    }
  }

#pragma unroll
  for (int u = 0; u < NT; ++u) {
    int n = n0 + 16 * u;
    float bb = bias[n];
#pragma unroll
    for (int v = 0; v < 8; ++v) {
      int m = tileM * 16 + v + (half << 3);
      float val = acc[u][v] + bb;
      if (resid) val += (float)resid[(size_t)m * Cout + n];
      if (reluFlag && val < 0.f) val = 0.f;
      if (mode == 0) {
        outb[(size_t)m * Cout + n] = (__bf16)val;
      } else {
        int b2 = m / Tout;
        int t2 = m - b2 * Tout;
        outf[((size_t)b2 * Cout + n) * Tout + t2] = val;   // NCH fp32
      }
    }
  }
}

// (B,C,T) fp32 -> (B,T,C) bf16
__global__ void x_to_btc_kernel(const float* __restrict__ x,
                                __bf16* __restrict__ out,
                                int Bn, int C, int T)
{
  size_t i = (size_t)blockIdx.x * blockDim.x + threadIdx.x;
  size_t total = (size_t)Bn * C * T;
  if (i >= total) return;
  int c = (int)(i % C);
  int t = (int)((i / C) % T);
  int b = (int)(i / ((size_t)C * T));
  out[i] = (__bf16)x[((size_t)b * C + c) * T + t];
}

// (O,I,Kw) fp32 -> (O, KtotPad) bf16 with kk = kw*I + i, zero pad to KtotPad
__global__ void prep_w_kernel(const float* __restrict__ w,
                              __bf16* __restrict__ wt,
                              int O, int I, int Kw, int KtotPad)
{
  size_t i = (size_t)blockIdx.x * blockDim.x + threadIdx.x;
  size_t total = (size_t)O * KtotPad;
  if (i >= total) return;
  int r = (int)(i % (size_t)KtotPad);
  int o = (int)(i / (size_t)KtotPad);
  float val = 0.f;
  if (r < I * Kw) {
    int kw = r / I;
    int ci = r - kw * I;
    val = w[((size_t)o * I + ci) * Kw + kw];
  }
  wt[i] = (__bf16)val;
}

__global__ void zero1_kernel(float* p) { *p = 0.f; }

__global__ void finalize_kernel(const float* accum, float* commit_p,
                                float* code_p, float denom)
{
  float v = *accum / denom;
  *commit_p = v;   // (flat-sel_un)^2 and (sel_un-flat)^2 are numerically equal
  *code_p   = v;
}

// Cosine-sim VQ: one wave per latent row (D=64, K=1024 codes).
__global__ __launch_bounds__(256) void vq_kernel(
    const __bf16* __restrict__ z,     // (N, 64) unnormalized latents (bf16)
    const float*  __restrict__ emb,   // (1024, 64) row-normalized codebook
    const float*  __restrict__ embUn, // (1024, 64) unnormalized codebook
    __bf16* __restrict__ q,           // (N, 64) selected normalized codes
    int*    __restrict__ idx_out,     // (N)
    float*  __restrict__ loss_accum)
{
  __shared__ float zrow[8][64];
  __shared__ float redv[8][32];
  __shared__ int   redi[8][32];
  int lane = threadIdx.x & 31;
  int wv   = threadIdx.x >> 5;
  int r    = blockIdx.x * 8 + wv;

  zrow[wv][lane]      = (float)z[(size_t)r * 64 + lane];
  zrow[wv][lane + 32] = (float)z[(size_t)r * 64 + lane + 32];
  __syncthreads();

  float sq = 0.f;
  for (int d = 0; d < 64; ++d) { float t = zrow[wv][d]; sq += t * t; }
  float rn = 1.0f / fmaxf(sqrtf(sq), 1e-12f);

  float bv = -1e30f; int bi = 0;
  for (int k = lane; k < 1024; k += 32) {
    const float* e = emb + (size_t)k * 64;
    float dot = 0.f;
    for (int d = 0; d < 64; ++d) dot += zrow[wv][d] * e[d];
    dot *= rn;
    if (dot > bv) { bv = dot; bi = k; }     // strict > keeps lowest k per lane
  }
  redv[wv][lane] = bv; redi[wv][lane] = bi;
  __syncthreads();
  if (lane == 0) {
    float mv = redv[wv][0]; int mi = redi[wv][0];
    for (int l = 1; l < 32; ++l) {
      float v = redv[wv][l]; int i = redi[wv][l];
      if (v > mv || (v == mv && i < mi)) { mv = v; mi = i; }
    }
    redi[wv][0] = mi;
  }
  __syncthreads();
  int sel = redi[wv][0];
  if (lane == 0) idx_out[r] = sel;

  const float* eN = emb   + (size_t)sel * 64;
  const float* eU = embUn + (size_t)sel * 64;
  q[(size_t)r * 64 + lane]      = (__bf16)eN[lane];
  q[(size_t)r * 64 + lane + 32] = (__bf16)eN[lane + 32];
  float d0 = zrow[wv][lane]      - eU[lane];
  float d1 = zrow[wv][lane + 32] - eU[lane + 32];
  redv[wv][lane] = d0 * d0 + d1 * d1;
  __syncthreads();
  if (lane == 0) {
    float s = 0.f;
    for (int l = 0; l < 32; ++l) s += redv[wv][l];
    atomicAdd(loss_accum, s);
  }
}

// ---------------------------------------------------------------------------
extern "C" void kernel_launch(void* const* d_in, const int* in_sizes, int n_in,
                              void* d_out, int out_size, void* d_ws, size_t ws_size,
                              hipStream_t stream)
{
  (void)in_sizes; (void)n_in; (void)out_size; (void)ws_size;
  const int B = 32, CIN = 16, T = 4096, H = 256, D = 64;
  const int T2 = T / 2, T4 = T / 4;

  float* dof      = (float*)d_out;
  float* x_recon  = dof;                          // B*CIN*T fp32 (NCH)
  float* commit_p = dof + (size_t)B * CIN * T;
  float* code_p   = commit_p + 1;
  int*   idx_p    = (int*)(code_p + 1);           // B*T4 int32

  size_t off = 0;
  auto carve = [&](size_t bytes) -> void* {
    void* p = (char*)d_ws + off;
    off += (bytes + 255) & ~(size_t)255;
    return p;
  };
  __bf16* F0 = (__bf16*)carve((size_t)B * T  * H   * 2);  // full act buf
  __bf16* F1 = (__bf16*)carve((size_t)B * T  * H   * 2);  // full act buf
  __bf16* Hh = (__bf16*)carve((size_t)B * T2 * H   * 2);  // half act buf
  __bf16* XB = (__bf16*)carve((size_t)B * T  * CIN * 2);
  __bf16* ZB = (__bf16*)carve((size_t)B * T4 * D   * 2);
  __bf16* QB = (__bf16*)carve((size_t)B * T4 * D   * 2);
  float*  accum = (float*)carve(256);

  // Conv layer table: d_in index of weight (bias at din+1), dims (O,I,Kw).
  struct LayerD { int din, O, I, Kw; };
  const LayerD L[33] = {
    {1,H,CIN,3},{3,H,H,4},
    {5,H,H,3},{7,H,H,3},{9,H,H,3},{11,H,H,3},{13,H,H,3},{15,H,H,3},   // enc l3
    {17,H,H,4},
    {19,H,H,3},{21,H,H,3},{23,H,H,3},{25,H,H,3},{27,H,H,3},{29,H,H,3},// enc l5
    {31,D,H,3},
    {33,H,D,3},
    {35,H,H,3},{37,H,H,3},{39,H,H,3},{41,H,H,3},{43,H,H,3},{45,H,H,3},// dec l2
    {47,H,H,3},
    {49,H,H,3},{51,H,H,3},{53,H,H,3},{55,H,H,3},{57,H,H,3},{59,H,H,3},// dec l4
    {61,H,H,3},{63,H,H,3},{65,CIN,H,3},
  };
  __bf16* WT[33];
  const float* BIAS[33];
  int KPAD[33];
  for (int i = 0; i < 33; ++i) {
    int ktot = L[i].I * L[i].Kw;
    KPAD[i] = (ktot + 31) & ~31;
    size_t ne = (size_t)L[i].O * KPAD[i];
    WT[i]   = (__bf16*)carve(ne * 2);
    BIAS[i] = (const float*)d_in[L[i].din + 1];
    int blk = (int)((ne + 255) / 256);
    prep_w_kernel<<<blk, 256, 0, stream>>>((const float*)d_in[L[i].din],
                                           WT[i], L[i].O, L[i].I, L[i].Kw,
                                           KPAD[i]);
  }
  const float* emb   = (const float*)d_in[67];
  const float* embUn = (const float*)d_in[68];

  {
    size_t ne = (size_t)B * T * CIN;
    x_to_btc_kernel<<<(int)((ne + 255) / 256), 256, 0, stream>>>(
        (const float*)d_in[0], XB, B, CIN, T);
  }
  zero1_kernel<<<1, 1, 0, stream>>>(accum);

  auto conv = [&](const __bf16* in, int TinStore, int TinEff, int halve,
                  int li, int Cin, int Cout, int stride, int pad, int dil,
                  const __bf16* resid, int relu, int Tout,
                  __bf16* outb, float* outf, int mode) {
    int Mtiles = (B * Tout) / 16;
    int cl2 = 31 - __builtin_clz((unsigned)Cin);
    if (Cout >= 32) {
      int Ntiles = Cout / 32;
      int waves  = Mtiles * Ntiles;
      int blocks = (waves + 7) / 8;
      conv1d_btc_wmma<2><<<blocks, 256, 0, stream>>>(
          in, WT[li], BIAS[li], resid, outb, outf,
          TinStore, TinEff, Tout, cl2, Cout, L[li].Kw, KPAD[li],
          stride, pad, dil, halve, relu, mode, Ntiles, waves);
    } else {
      int Ntiles = Cout / 16;
      int waves  = Mtiles * Ntiles;
      int blocks = (waves + 7) / 8;
      conv1d_btc_wmma<1><<<blocks, 256, 0, stream>>>(
          in, WT[li], BIAS[li], resid, outb, outf,
          TinStore, TinEff, Tout, cl2, Cout, L[li].Kw, KPAD[li],
          stride, pad, dil, halve, relu, mode, Ntiles, waves);
    }
  };

  const int DILS[3] = {1, 3, 9};
  __bf16 *cur, *tmpb;

  // ---------------- encoder ----------------
  conv(XB, T, T, 0, 0, CIN, H, 1, 1, 1, nullptr, 1, T, F0, nullptr, 0);
  conv(F0, T, T, 0, 1, H, H, 2, 1, 1, nullptr, 0, T2, F1, nullptr, 0);
  cur = F1; tmpb = Hh;                              // F0 = c1 scratch
  for (int j = 0; j < 3; ++j) {
    int d = DILS[j];
    conv(cur, T2, T2, 0, 2 + 2 * j, H, H, 1, d, d, nullptr, 1, T2, F0, nullptr, 0);
    conv(F0,  T2, T2, 0, 3 + 2 * j, H, H, 1, d, d, cur,     1, T2, tmpb, nullptr, 0);
    __bf16* t = cur; cur = tmpb; tmpb = t;
  }                                                 // cur == Hh
  conv(cur, T2, T2, 0, 8, H, H, 2, 1, 1, nullptr, 0, T4, F0, nullptr, 0);
  cur = F0; tmpb = Hh;                              // F1 = c1 scratch
  for (int j = 0; j < 3; ++j) {
    int d = DILS[j];
    conv(cur, T4, T4, 0, 9  + 2 * j, H, H, 1, d, d, nullptr, 1, T4, F1, nullptr, 0);
    conv(F1,  T4, T4, 0, 10 + 2 * j, H, H, 1, d, d, cur,     1, T4, tmpb, nullptr, 0);
    __bf16* t = cur; cur = tmpb; tmpb = t;
  }                                                 // cur == Hh
  conv(cur, T4, T4, 0, 15, H, D, 1, 1, 1, nullptr, 0, T4, ZB, nullptr, 0);

  // ---------------- vector quantize ----------------
  vq_kernel<<<(B * T4) / 8, 256, 0, stream>>>(ZB, emb, embUn, QB, idx_p, accum);
  finalize_kernel<<<1, 1, 0, stream>>>(accum, commit_p, code_p,
                                       (float)((size_t)B * T4 * D));

  // ---------------- decoder ----------------
  conv(QB, T4, T4, 0, 16, D, H, 1, 1, 1, nullptr, 1, T4, F0, nullptr, 0);
  cur = F0; tmpb = Hh;
  for (int j = 0; j < 3; ++j) {
    int d = DILS[j];
    conv(cur, T4, T4, 0, 17 + 2 * j, H, H, 1, d, d, nullptr, 1, T4, F1, nullptr, 0);
    conv(F1,  T4, T4, 0, 18 + 2 * j, H, H, 1, d, d, cur,     1, T4, tmpb, nullptr, 0);
    __bf16* t = cur; cur = tmpb; tmpb = t;
  }                                                 // cur == Hh
  conv(cur, T4, T2, 1, 23, H, H, 1, 1, 1, nullptr, 0, T2, F0, nullptr, 0); // up x2
  cur = F0; tmpb = Hh;
  for (int j = 0; j < 3; ++j) {
    int d = DILS[j];
    conv(cur, T2, T2, 0, 24 + 2 * j, H, H, 1, d, d, nullptr, 1, T2, F1, nullptr, 0);
    conv(F1,  T2, T2, 0, 25 + 2 * j, H, H, 1, d, d, cur,     1, T2, tmpb, nullptr, 0);
    __bf16* t = cur; cur = tmpb; tmpb = t;
  }                                                 // cur == Hh
  conv(cur, T2, T, 1, 30, H, H, 1, 1, 1, nullptr, 1, T, F0, nullptr, 0);   // up x2 + relu
  conv(F0,  T, T, 0, 31, H, H, 1, 1, 1, nullptr, 1, T, F1, nullptr, 0);    // l6a
  conv(F1,  T, T, 0, 32, H, CIN, 1, 1, 1, nullptr, 0, T, nullptr, x_recon, 1); // l6b -> NCH
}
